// MultiHeadAttention_61237643707039
// MI455X (gfx1250) — compile-verified, matrix operand order checked by reading, not run on hardware
//
#include <hip/hip_runtime.h>
#include <stdint.h>

// ---------------- problem constants ----------------
#define BATCH 2
#define S_LEN 2048
#define DM    1024
#define NH    16
#define DK    64

typedef __attribute__((ext_vector_type(16))) __bf16 v16bf;
typedef __attribute__((ext_vector_type(8)))  float  v8f;
typedef __attribute__((ext_vector_type(4)))  float  v4f;
typedef unsigned int v4u __attribute__((ext_vector_type(4)));
typedef int v8i_t __attribute__((ext_vector_type(8)));
typedef int v4i_t __attribute__((ext_vector_type(4)));

// fp32 -> bf16 round-to-nearest-even
__device__ __forceinline__ unsigned short f2bf(float f) {
  unsigned int u = __float_as_uint(f);
  unsigned int r = u + 0x7FFFu + ((u >> 16) & 1u);
  return (unsigned short)(r >> 16);
}

// Load a 16x32 bf16 WMMA operand fragment from a row-major bf16 matrix.
// ISA layout (16-bit A, 16x32): lane r (0-15) holds row r; lanes 0-15 carry
// K = {0..7, 16..23}, lanes 16-31 carry K = {8..15, 24..31}. B operand uses
// the mirrored layout with lane = N, so the same loader reads B^T row-major.
__device__ __forceinline__ v16bf load_frag_rm(const unsigned short* src,
                                              int row0, int col0, int ld,
                                              int r, int hi) {
  const unsigned short* p = src + (size_t)(row0 + r) * ld + col0 + hi * 8;
  v16bf f;
  uint4* pf = (uint4*)&f;
  pf[0] = *(const uint4*)(p);
  pf[1] = *(const uint4*)(p + 16);
  return f;
}

// Same fragment, but sourced from fp32 (converted in registers).
__device__ __forceinline__ v16bf load_frag_f32(const float* src,
                                               int row0, int col0, int ld,
                                               int r, int hi) {
  const float* p = src + (size_t)(row0 + r) * ld + col0 + hi * 8;
  v16bf f;
  unsigned short* u = (unsigned short*)&f;
#pragma unroll
  for (int i = 0; i < 8; ++i) u[i] = f2bf(p[i]);
#pragma unroll
  for (int i = 0; i < 8; ++i) u[8 + i] = f2bf(p[16 + i]);
  return f;
}

__device__ __forceinline__ v8f wmma_bf16(v16bf a, v16bf b, v8f c) {
  return __builtin_amdgcn_wmma_f32_16x16x32_bf16(false, a, false, b,
                                                 (short)0, c, false, false);
}

// LDS byte offset of a __shared__ pointer: per ISA, the low 32 bits of the
// generic (flat) address are the wave-relative LDS offset.
__device__ __forceinline__ unsigned int lds_off(const void* p) {
  return (unsigned int)(unsigned long long)p;
}

// Issue a TDM 2D tile load (bf16 elements) global -> LDS.
// D# built per CDNA5 ISA 8.3/8.4: group0 = {count, lds_addr, global_addr,
// type=2}; group1 = {data_size=1(2B), tensor dims, tile dims, dim0 stride}.
// This toolchain's builtin takes 6 args (g0, g1, g2, g3, extra group, cpol).
__device__ __forceinline__ void tdm_load_2d_bf16(unsigned int ldsaddr,
                                                 const void* gptr,
                                                 unsigned int tensor_d0,
                                                 unsigned int tensor_d1,
                                                 unsigned int tile_d0,
                                                 unsigned int tile_d1,
                                                 unsigned int stride0) {
  unsigned long long ga = (unsigned long long)gptr;
  v4u g0;
  g0.x = 1u;                                              // count=1 (valid)
  g0.y = ldsaddr;                                         // lds_addr bytes
  g0.z = (unsigned int)(ga & 0xFFFFFFFFull);              // global_addr[31:0]
  g0.w = (unsigned int)((ga >> 32) & 0x1FFFFFFull) | (2u << 30); // [56:32]|type=2
  v8i_t g1;
  g1[0] = (int)(1u << 16);                                // data_size = 2 bytes
  g1[1] = (int)((tensor_d0 & 0xFFFFu) << 16);             // tensor_dim0 lo
  g1[2] = (int)((tensor_d0 >> 16) | ((tensor_d1 & 0xFFFFu) << 16));
  g1[3] = (int)((tensor_d1 >> 16) | (tile_d0 << 16));     // tile_dim0
  g1[4] = (int)(tile_d1 & 0xFFFFu);                       // tile_dim1 (dim2=0)
  g1[5] = (int)stride0;                                   // dim0 stride lo32
  g1[6] = 0;
  g1[7] = 0;
  v4i_t z4 = {0, 0, 0, 0};                                // 2D: groups 2/3 unused
  v8i_t z8 = {0, 0, 0, 0, 0, 0, 0, 0};
  __builtin_amdgcn_tensor_load_to_lds(g0, g1, z4, z4, z8, 0);
}

// ---------------- kernel 1: weight transpose + bf16 convert ----------------
__global__ void mha_wtrans_kernel(const float* __restrict__ W,
                                  unsigned short* __restrict__ Wt) {
  int idx = blockIdx.x * blockDim.x + threadIdx.x;
  int n = idx & (DM - 1);
  int k = idx >> 10;
  Wt[(size_t)n * DM + k] = f2bf(W[idx]);
}

// ---------------- kernel 2: Q/K/V projection GEMM ----------------
template <int VMODE>
__global__ void mha_proj_kernel(const float* __restrict__ X,
                                const unsigned short* __restrict__ Wt,
                                unsigned short* __restrict__ out) {
  int wid  = (blockIdx.x * blockDim.x + threadIdx.x) >> 5;
  int lane = threadIdx.x & 31;
  int r = lane & 15, hi = lane >> 4;
  int mt = wid >> 4, ng = wid & 15;
  int m0 = mt * 16, n0 = ng * 64;

  v8f acc[4] = {};
  for (int k0 = 0; k0 < DM; k0 += 32) {
    __builtin_prefetch(X + (size_t)(m0 + r) * DM + k0 + 32, 0, 0);
    v16bf a = load_frag_f32(X, m0, k0, DM, r, hi);
#pragma unroll
    for (int i = 0; i < 4; ++i) {
      v16bf b = load_frag_rm(Wt, n0 + i * 16, k0, DM, r, hi);
      acc[i] = wmma_bf16(a, b, acc[i]);
    }
  }
#pragma unroll
  for (int i = 0; i < 4; ++i) {
#pragma unroll
    for (int v = 0; v < 8; ++v) {
      int m = m0 + v + hi * 8;
      int n = n0 + i * 16 + r;
      unsigned short val = f2bf(acc[i][v]);
      if (VMODE == 0) {
        out[(size_t)m * DM + n] = val;
      } else {
        int b = m >> 11, s = m & (S_LEN - 1);
        int h = n >> 6,  d = n & 63;
        out[((size_t)(b * NH + h) * DK + d) * S_LEN + s] = val;
      }
    }
  }
}

// ---------------- kernel 3: attention ----------------
// One block = one (b,h) x 4 consecutive q-tiles (64 q rows). K/V tiles are
// streamed into LDS double buffers by the Tensor Data Mover (wave 0 issues,
// s_wait_tensorcnt + barrier publishes), shared by all 4 waves. All waves
// run a uniform chunk count; fully-masked chunks produce exact zero probs,
// which the materialized attn output needs anyway.
__global__ void __launch_bounds__(128)
mha_attn_kernel(const unsigned short* __restrict__ Qb,
                const unsigned short* __restrict__ Kb,
                const unsigned short* __restrict__ Vt,
                unsigned short* __restrict__ Ctx,
                float* __restrict__ attn_out) {
  __shared__ __align__(16) unsigned short Kst[2][32 * 64];  // 8 KB
  __shared__ __align__(16) unsigned short Vst[2][64 * 32];  // 8 KB
  __shared__ __align__(16) unsigned short pstage[4][16 * 32];

  int wIn  = threadIdx.x >> 5;
  int lane = threadIdx.x & 31;
  int r = lane & 15, hi = lane >> 4;

  int bh   = blockIdx.x >> 5;         // 0..31 = b*16 + h
  int qblk = blockIdx.x & 31;         // 32 q-blocks of 64 rows
  int b = bh >> 4, h = bh & 15;
  int qt = qblk * 4 + wIn;
  int q0 = qt * 16;
  int nch = (qblk + 1) * 2;           // 32-key chunks covering [0, (qblk+1)*64)

  const unsigned short* Qsrc = Qb + (size_t)b * S_LEN * DM;
  const unsigned short* Ksrc = Kb + (size_t)b * S_LEN * DM;   // ld = DM
  const unsigned short* Vsrc = Vt + (size_t)bh * DK * S_LEN;  // [64,S], ld = S
  float* arow = attn_out + (size_t)bh * S_LEN * S_LEN;

  bool issuer = (wIn == 0);
  unsigned int koff[2] = { lds_off(Kst[0]), lds_off(Kst[1]) };
  unsigned int voff[2] = { lds_off(Vst[0]), lds_off(Vst[1]) };

  v16bf qa0 = load_frag_rm(Qsrc, q0, h * DK,      DM, r, hi);
  v16bf qa1 = load_frag_rm(Qsrc, q0, h * DK + 32, DM, r, hi);

  const float scale = 0.125f;
  float mrow[8], lrow[8];
#pragma unroll
  for (int v = 0; v < 8; ++v) { mrow[v] = -3.0e38f; lrow[v] = 0.0f; }

  // ---- pass 1: online softmax statistics (K tiles via TDM) ----
  if (issuer)
    tdm_load_2d_bf16(koff[0], Ksrc + h * DK, DM, S_LEN, 64, 32, DM);
  for (int i = 0; i < nch; ++i) {
    int k0 = i * 32;
    if (issuer) {
      if (i + 1 < nch) {
        tdm_load_2d_bf16(koff[(i + 1) & 1],
                         Ksrc + (size_t)(k0 + 32) * DM + h * DK,
                         DM, S_LEN, 64, 32, DM);
        __builtin_amdgcn_s_wait_tensorcnt(1);   // current chunk landed
      } else {
        __builtin_amdgcn_s_wait_tensorcnt(0);
      }
    }
    __syncthreads();
    const unsigned short* kt = Kst[i & 1];      // [32 keys][64 dims]
    v8f s[2] = {};
#pragma unroll
    for (int t = 0; t < 2; ++t) {
      v16bf kb0 = load_frag_rm(kt, t * 16, 0,  64, r, hi);
      v16bf kb1 = load_frag_rm(kt, t * 16, 32, 64, r, hi);
      s[t] = wmma_bf16(qa0, kb0, s[t]);
      s[t] = wmma_bf16(qa1, kb1, s[t]);
    }
#pragma unroll
    for (int v = 0; v < 8; ++v) {
      int m = v + hi * 8;
      float s0 = s[0][v] * scale;
      float s1 = s[1][v] * scale;
      if (k0 + r      > q0 + m) s0 = -1e30f;
      if (k0 + 16 + r > q0 + m) s1 = -1e30f;
      float cm = fmaxf(s0, s1);
#pragma unroll
      for (int off = 1; off < 16; off <<= 1) cm = fmaxf(cm, __shfl_xor(cm, off, 16));
      float nm = fmaxf(mrow[v], cm);
      float e  = __expf(s0 - nm) + __expf(s1 - nm);
#pragma unroll
      for (int off = 1; off < 16; off <<= 1) e += __shfl_xor(e, off, 16);
      lrow[v] = lrow[v] * __expf(mrow[v] - nm) + e;
      mrow[v] = nm;
    }
    __syncthreads();                            // buffer reuse fence
  }

  float linv[8];
#pragma unroll
  for (int v = 0; v < 8; ++v) linv[v] = 1.0f / lrow[v];

  // ---- pass 2: probs out (non-temporal) + context accumulate ----
  v8f acc[4] = {};
  unsigned short* myst = pstage[wIn];
  if (issuer) {
    tdm_load_2d_bf16(koff[0], Ksrc + h * DK, DM, S_LEN, 64, 32, DM);
    tdm_load_2d_bf16(voff[0], Vsrc, S_LEN, DK, 32, 64, S_LEN);
  }
  for (int i = 0; i < nch; ++i) {
    int k0 = i * 32;
    if (issuer) {
      if (i + 1 < nch) {
        tdm_load_2d_bf16(koff[(i + 1) & 1],
                         Ksrc + (size_t)(k0 + 32) * DM + h * DK,
                         DM, S_LEN, 64, 32, DM);
        tdm_load_2d_bf16(voff[(i + 1) & 1], Vsrc + k0 + 32,
                         S_LEN, DK, 32, 64, S_LEN);
        __builtin_amdgcn_s_wait_tensorcnt(2);
      } else {
        __builtin_amdgcn_s_wait_tensorcnt(0);
      }
    }
    __syncthreads();
    const unsigned short* kt = Kst[i & 1];
    const unsigned short* vt = Vst[i & 1];      // [64 dims][32 keys]
    v8f s[2] = {};
#pragma unroll
    for (int t = 0; t < 2; ++t) {
      v16bf kb0 = load_frag_rm(kt, t * 16, 0,  64, r, hi);
      v16bf kb1 = load_frag_rm(kt, t * 16, 32, 64, r, hi);
      s[t] = wmma_bf16(qa0, kb0, s[t]);
      s[t] = wmma_bf16(qa1, kb1, s[t]);
    }
#pragma unroll
    for (int t = 0; t < 2; ++t) {
#pragma unroll
      for (int v = 0; v < 8; ++v) {
        int m = v + hi * 8;
        float sv = s[t][v] * scale;
        if (k0 + t * 16 + r > q0 + m) sv = -1e30f;
        float p = __expf(sv - mrow[v]) * linv[v];
        // streaming 512MB output: NT so K/V stay L2-resident
        __builtin_nontemporal_store(p, arow + (size_t)(q0 + m) * S_LEN + k0 + t * 16 + r);
        myst[m * 32 + t * 16 + r] = f2bf(p);    // D-layout -> LDS
      }
    }
    asm volatile("s_wait_dscnt 0" ::: "memory"); // cross-lane LDS handoff
    v16bf pa = load_frag_rm(myst, 0, 0, 32, r, hi);
#pragma unroll
    for (int i2 = 0; i2 < 4; ++i2) {
      v16bf vb = load_frag_rm(vt, i2 * 16, 0, 32, r, hi);
      acc[i2] = wmma_bf16(pa, vb, acc[i2]);
    }
    __syncthreads();
  }

  // store context [B, S, H*64] bf16
#pragma unroll
  for (int i = 0; i < 4; ++i) {
#pragma unroll
    for (int v = 0; v < 8; ++v) {
      int m = q0 + v + hi * 8;
      int n = h * DK + i * 16 + r;
      Ctx[((size_t)b * S_LEN + m) * DM + n] = f2bf(acc[i][v]);
    }
  }

  // zero-fill the fully-masked causal tail (chunks already wrote exact zeros
  // up to the block's processed end)
  int kProcEnd = (qblk + 1) * 64;
  v4f z = {0.f, 0.f, 0.f, 0.f};
  for (int m = 0; m < 16; ++m) {
    float* rowp = arow + (size_t)(q0 + m) * S_LEN;
    for (int c = kProcEnd + lane * 4; c < S_LEN; c += 128)
      __builtin_nontemporal_store(z, (v4f*)(rowp + c));
  }
}

// ---------------- kernel 4: out-proj + residual + LayerNorm (fused) -------
__global__ void __launch_bounds__(512)
mha_oproj_ln_kernel(const unsigned short* __restrict__ Ctx,
                    const unsigned short* __restrict__ WoT,
                    const float* __restrict__ resid,
                    const float* __restrict__ gamma,
                    const float* __restrict__ beta,
                    float* __restrict__ out) {
  __shared__ __align__(16) float ybuf[16][DM];   // 64 KB of the 320 KB LDS
  int m0   = blockIdx.x * 16;
  int w    = threadIdx.x >> 5;
  int lane = threadIdx.x & 31;
  int r = lane & 15, hi = lane >> 4;
  int n0 = w * 64;

  v8f acc[4] = {};
  for (int k0 = 0; k0 < DM; k0 += 32) {
    v16bf a = load_frag_rm(Ctx, m0, k0, DM, r, hi);
#pragma unroll
    for (int i = 0; i < 4; ++i) {
      v16bf bfrg = load_frag_rm(WoT, n0 + i * 16, k0, DM, r, hi);
      acc[i] = wmma_bf16(a, bfrg, acc[i]);
    }
  }
#pragma unroll
  for (int i = 0; i < 4; ++i) {
#pragma unroll
    for (int v = 0; v < 8; ++v) {
      int m = v + hi * 8;
      int n = n0 + i * 16 + r;
      ybuf[m][n] = acc[i][v] + resid[(size_t)(m0 + m) * DM + n];
    }
  }
  __syncthreads();

  float sum = 0.f, sumsq = 0.f;
  for (int c = lane; c < DM; c += 32) {
    float x = ybuf[w][c];
    sum += x; sumsq += x * x;
  }
#pragma unroll
  for (int off = 1; off < 32; off <<= 1) {
    sum   += __shfl_xor(sum,   off, 32);
    sumsq += __shfl_xor(sumsq, off, 32);
  }
  float mu   = sum * (1.0f / DM);
  float var  = sumsq * (1.0f / DM) - mu * mu;
  float rstd = rsqrtf(var + 1e-5f);
  float* orow = out + (size_t)(m0 + w) * DM;
  for (int c = lane; c < DM; c += 32)
    orow[c] = (ybuf[w][c] - mu) * rstd * gamma[c] + beta[c];
}

// ---------------- launch ----------------
extern "C" void kernel_launch(void* const* d_in, const int* in_sizes, int n_in,
                              void* d_out, int out_size, void* d_ws, size_t ws_size,
                              hipStream_t stream) {
  const float* inQ   = (const float*)d_in[0];
  const float* inK   = (const float*)d_in[1];
  const float* inV   = (const float*)d_in[2];
  // d_in[3] = attn_mask (causal, computed analytically)
  const float* WQ    = (const float*)d_in[4];
  const float* WK    = (const float*)d_in[5];
  const float* WV    = (const float*)d_in[6];
  const float* WO    = (const float*)d_in[7];
  const float* gamma = (const float*)d_in[8];
  const float* beta  = (const float*)d_in[9];

  float* out  = (float*)d_out;
  float* attn = out + (size_t)BATCH * S_LEN * DM;

  unsigned short* WqT = (unsigned short*)d_ws;
  unsigned short* WkT = WqT + (size_t)DM * DM;
  unsigned short* WvT = WkT + (size_t)DM * DM;
  unsigned short* WoT = WvT + (size_t)DM * DM;
  unsigned short* Qb  = WoT + (size_t)DM * DM;
  unsigned short* Kb  = Qb  + (size_t)BATCH * S_LEN * DM;
  unsigned short* Vt  = Kb  + (size_t)BATCH * S_LEN * DM;
  unsigned short* Ctx = Vt  + (size_t)BATCH * S_LEN * DM;

  int wblk = (DM * DM) / 256;
  mha_wtrans_kernel<<<wblk, 256, 0, stream>>>(WQ, WqT);
  mha_wtrans_kernel<<<wblk, 256, 0, stream>>>(WK, WkT);
  mha_wtrans_kernel<<<wblk, 256, 0, stream>>>(WV, WvT);
  mha_wtrans_kernel<<<wblk, 256, 0, stream>>>(WO, WoT);

  mha_proj_kernel<0><<<1024, 128, 0, stream>>>(inQ, WqT, Qb);
  mha_proj_kernel<0><<<1024, 128, 0, stream>>>(inK, WkT, Kb);
  mha_proj_kernel<1><<<1024, 128, 0, stream>>>(inV, WvT, Vt);

  // 1024 blocks = (B*H=32) x (32 q-blocks of 64 rows), 4 waves each
  mha_attn_kernel<<<1024, 128, 0, stream>>>(Qb, Kb, Vt, Ctx, attn);

  mha_oproj_ln_kernel<<<256, 512, 0, stream>>>(Ctx, WoT, inQ, gamma, beta, out);
}
